// AMRBERTforClassification_45268955299903
// MI455X (gfx1250) — compile-verified, matrix operand order checked by reading, not run on hardware
//
#include <hip/hip_runtime.h>
#include <hip/hip_bf16.h>

typedef __bf16 bf16;
typedef __attribute__((ext_vector_type(8)))  __bf16 bf16x8;
typedef __attribute__((ext_vector_type(16))) __bf16 v16bf;
typedef __attribute__((ext_vector_type(8)))  float  v8f;

#define D_IN   768
#define D_OUT  256
#define KTILES (D_IN / 32)    // 24 K-steps of 32
#define NTILES (D_OUT / 16)   // 16 N-tiles of 16
#define NGRAPH 16
#define NEG_SLOPE 0.2f
#define LDA 776               // bf16 LDS row stride (768 + 8), row bytes = 1552 (16B multiple)

// ---- workspace layout (bytes) ----
#define OFF_BL     0x000000u          // packed Wl  (768*256*2 = 384 KB)
#define OFF_BR     0x060000u          // packed Wr
#define OFF_E      0x100000u          // edge logits / exp, (E+N)*4 = 1.25 MB
#define OFF_MKEY   0x240000u          // per-node max key (u32), 256 KB
#define OFF_DENOM  0x280000u          // per-node softmax denom, 256 KB
#define OFF_POOLED 0x2C0000u          // 16*256 f32
#define OFF_CNT    0x2C4000u          // 16 f32
#define OFF_EMB_P  0x2C8000u          // 16*256 f32
#define OFF_EMB_H  0x2CC000u          // 16*256 f32
#define OFF_XL     0x0400000u         // 64 MB
#define OFF_XR     0x4400000u         // 64 MB (reused as aggregation target h)

// ---------- helpers ----------
__device__ inline bf16 f2bf(float f) {
  unsigned u = __float_as_uint(f);
  u += 0x7FFFu + ((u >> 16) & 1u);          // round-to-nearest-even
  unsigned short s = (unsigned short)(u >> 16);
  return __builtin_bit_cast(bf16, s);
}
__device__ inline unsigned f2key(float f) { // order-preserving float->u32 key
  unsigned u = __float_as_uint(f);
  return (u & 0x80000000u) ? ~u : (u | 0x80000000u);
}
__device__ inline float key2f(unsigned k) {
  unsigned u = (k & 0x80000000u) ? (k & 0x7FFFFFFFu) : ~k;
  return __uint_as_float(u);
}
__device__ inline void atomAddF(float* p, float v) {
  __hip_atomic_fetch_add(p, v, __ATOMIC_RELAXED, __HIP_MEMORY_SCOPE_AGENT); // global_atomic_add_f32
}

// ---------- pack weights into per-lane WMMA B-fragment order ----------
// Bpack[((nt*KTILES + t)*32 + lane)*16 + j]; lane 0-15: n=lane, K {j<8: t*32+j, else t*32+16+j-8}
// lane 16-31: n=lane-16, K {j<8: t*32+8+j, else t*32+24+j-8}
__global__ void pack_w_kernel(const float* __restrict__ Wl, const float* __restrict__ Wr,
                              bf16* __restrict__ Bl, bf16* __restrict__ Br) {
  int o = blockIdx.x * blockDim.x + threadIdx.x;
  if (o >= KTILES * NTILES * 512) return;
  int j    = o & 15;
  int lane = (o >> 4) & 31;
  int t    = (o >> 9) % KTILES;
  int nt   = o / (KTILES * 512);
  int khalf = (lane < 16) ? 0 : 8;
  int k = t * 32 + ((j < 8) ? (khalf + j) : (khalf + 16 + (j - 8)));
  int n = nt * 16 + (lane & 15);
  Bl[o] = f2bf(Wl[k * D_OUT + n]);
  Br[o] = f2bf(Wr[k * D_OUT + n]);
}

// ---------- fused GEMM: xl = x@Wl + bl, xr = x@Wr + br (bf16 WMMA, f32 accum) ----------
__global__ __launch_bounds__(512)
void gemm_xlxr_kernel(const float* __restrict__ x,
                      const bf16* __restrict__ Bl, const bf16* __restrict__ Br,
                      const float* __restrict__ blv, const float* __restrict__ brv,
                      float* __restrict__ xl, float* __restrict__ xr) {
  __shared__ bf16 As[16 * LDA];
  const int tid = threadIdx.x;
  const long rowTile = blockIdx.x;                       // 16 rows of x per block
  const float* xrow = x + rowTile * 16 * D_IN;

  // stage 16x768 f32 tile -> bf16 LDS
  for (int f4 = tid; f4 < (16 * D_IN) / 4; f4 += 512) {
    int r  = f4 / (D_IN / 4);
    int c4 = f4 % (D_IN / 4);
    float4 v = ((const float4*)xrow)[f4];
    bf16* dst = &As[r * LDA + c4 * 4];
    dst[0] = f2bf(v.x); dst[1] = f2bf(v.y); dst[2] = f2bf(v.z); dst[3] = f2bf(v.w);
  }
  __syncthreads();

  const int wave  = tid >> 5;            // n-tile index (16 waves = 16 n-tiles)
  const int lane  = tid & 31;
  const int row   = lane & 15;
  const int khalf = (lane < 16) ? 0 : 8;

  v8f cl = {}; v8f cr = {};
  const bf16* bpl = Bl + wave * KTILES * 512 + lane * 16;
  const bf16* bpr = Br + wave * KTILES * 512 + lane * 16;

  for (int t = 0; t < KTILES; ++t) {
    const bf16* ap = &As[row * LDA + t * 32 + khalf];
    bf16x8 a_lo = *(const bf16x8*)(ap);                  // ds_load_b128
    bf16x8 a_hi = *(const bf16x8*)(ap + 16);
    v16bf a = __builtin_shufflevector(a_lo, a_hi, 0,1,2,3,4,5,6,7,8,9,10,11,12,13,14,15);

    const bf16* pl = bpl + t * 512;
    __builtin_prefetch(pl + 512, 0, 3);                  // WGP-scope prefetch of next K-step (high reuse)
    bf16x8 l0 = *(const bf16x8*)(pl);
    bf16x8 l1 = *(const bf16x8*)(pl + 8);
    v16bf bl16 = __builtin_shufflevector(l0, l1, 0,1,2,3,4,5,6,7,8,9,10,11,12,13,14,15);
    cl = __builtin_amdgcn_wmma_f32_16x16x32_bf16(false, a, false, bl16, (short)0, cl, false, false);

    const bf16* pr = bpr + t * 512;
    bf16x8 r0 = *(const bf16x8*)(pr);
    bf16x8 r1 = *(const bf16x8*)(pr + 8);
    v16bf br16 = __builtin_shufflevector(r0, r1, 0,1,2,3,4,5,6,7,8,9,10,11,12,13,14,15);
    cr = __builtin_amdgcn_wmma_f32_16x16x32_bf16(false, a, false, br16, (short)0, cr, false, false);
  }

  // D layout: VGPR r -> M=r (lanes 0-15) / M=r+8 (lanes 16-31); N = lane&15
  const int n = wave * 16 + (lane & 15);
  const long mbase = rowTile * 16 + ((lane < 16) ? 0 : 8);
  const float bL = blv[n], bR = brv[n];
#pragma unroll
  for (int r8 = 0; r8 < 8; ++r8) {
    xl[(mbase + r8) * D_OUT + n] = cl[r8] + bL;
    xr[(mbase + r8) * D_OUT + n] = cr[r8] + bR;
  }
}

// ---------- edge attention logits + segment-max (one wave32 per edge) ----------
__global__ __launch_bounds__(256)
void edge_logits_kernel(const int* __restrict__ ei, int n_edges, int n_nodes,
                        const float* __restrict__ xl, const float* __restrict__ xr,
                        const float* __restrict__ att,
                        float* __restrict__ e_out, unsigned* __restrict__ mkey) {
  int eid = blockIdx.x * 8 + (threadIdx.x >> 5);
  int tot = n_edges + n_nodes;
  if (eid >= tot) return;
  int lane = threadIdx.x & 31;
  int src, dst;
  if (eid < n_edges) { src = ei[eid]; dst = ei[n_edges + eid]; }
  else               { src = dst = eid - n_edges; }          // self loop

  const float4* pl = (const float4*)(xl + (long)src * D_OUT) + lane * 2;
  const float4* pr = (const float4*)(xr + (long)dst * D_OUT) + lane * 2;
  const float4* pa = (const float4*)att + lane * 2;
  float acc = 0.f;
#pragma unroll
  for (int h = 0; h < 2; ++h) {
    float4 a = pl[h], b = pr[h], w = pa[h];
    float s;
    s = a.x + b.x; acc += (s > 0.f ? s : NEG_SLOPE * s) * w.x;
    s = a.y + b.y; acc += (s > 0.f ? s : NEG_SLOPE * s) * w.y;
    s = a.z + b.z; acc += (s > 0.f ? s : NEG_SLOPE * s) * w.z;
    s = a.w + b.w; acc += (s > 0.f ? s : NEG_SLOPE * s) * w.w;
  }
#pragma unroll
  for (int off = 16; off; off >>= 1) acc += __shfl_xor(acc, off, 32);
  if (lane == 0) {
    e_out[eid] = acc;
    atomicMax(&mkey[dst], f2key(acc));                   // global_atomic_max_u32
  }
}

// ---------- exp(e - max) + segment-sum denominator ----------
__global__ void edge_exp_kernel(const int* __restrict__ ei, int n_edges, int n_nodes,
                                const unsigned* __restrict__ mkey,
                                float* __restrict__ e_arr, float* __restrict__ denom) {
  int eid = blockIdx.x * blockDim.x + threadIdx.x;
  int tot = n_edges + n_nodes;
  if (eid >= tot) return;
  int dst = (eid < n_edges) ? ei[n_edges + eid] : (eid - n_edges);
  float ex = __expf(e_arr[eid] - key2f(mkey[dst]));
  e_arr[eid] = ex;
  atomAddF(&denom[dst], ex);
}

// ---------- h[dst] += alpha * xl[src]  (wave32 per edge, f32 atomics) ----------
__global__ __launch_bounds__(256)
void scatter_kernel(const int* __restrict__ ei, int n_edges, int n_nodes,
                    const float* __restrict__ e_arr, const float* __restrict__ denom,
                    const float* __restrict__ xl, float* __restrict__ h) {
  int eid = blockIdx.x * 8 + (threadIdx.x >> 5);
  int tot = n_edges + n_nodes;
  if (eid >= tot) return;
  int lane = threadIdx.x & 31;
  int src, dst;
  if (eid < n_edges) { src = ei[eid]; dst = ei[n_edges + eid]; }
  else               { src = dst = eid - n_edges; }
  float alpha = e_arr[eid] / denom[dst];
  const float* ps = xl + (long)src * D_OUT + lane * 8;
  float*       pd = h  + (long)dst * D_OUT + lane * 8;
  float4 v0 = *(const float4*)ps;
  float4 v1 = *(const float4*)(ps + 4);
  atomAddF(pd + 0, alpha * v0.x); atomAddF(pd + 1, alpha * v0.y);
  atomAddF(pd + 2, alpha * v0.z); atomAddF(pd + 3, alpha * v0.w);
  atomAddF(pd + 4, alpha * v1.x); atomAddF(pd + 5, alpha * v1.y);
  atomAddF(pd + 6, alpha * v1.z); atomAddF(pd + 7, alpha * v1.w);
}

// ---------- zero / init ----------
__global__ void fill4_kernel(float4* p, int n4) {
  int i = blockIdx.x * blockDim.x + threadIdx.x;
  if (i < n4) p[i] = make_float4(0.f, 0.f, 0.f, 0.f);
}
__global__ void init_side_kernel(unsigned* mkey, float* denom, float* pooled, float* cnt, int n_nodes) {
  int i = blockIdx.x * blockDim.x + threadIdx.x;
  if (i < n_nodes) { mkey[i] = 0u; denom[i] = 0.f; }
  if (i < NGRAPH * D_OUT) pooled[i] = 0.f;
  if (i < NGRAPH) cnt[i] = 0.f;
}

// ---------- mean pool (wave32 per node) ----------
__global__ __launch_bounds__(256)
void pool_kernel(const float* __restrict__ h, const int* __restrict__ batch,
                 float* __restrict__ pooled, float* __restrict__ cnt, int n_nodes) {
  int node = blockIdx.x * 8 + (threadIdx.x >> 5);
  if (node >= n_nodes) return;
  int lane = threadIdx.x & 31;
  int b = batch[node];
  const float* ph = h + (long)node * D_OUT + lane * 8;
  float*       pp = pooled + b * D_OUT + lane * 8;
#pragma unroll
  for (int j = 0; j < 8; ++j) atomAddF(pp + j, ph[j]);
  if (lane == 0) atomAddF(&cnt[b], 1.0f);
}

__global__ void finalize_pool_kernel(const float* pooled, const float* cnt,
                                     const float* gbias, float* emb) {
  int i = blockIdx.x * blockDim.x + threadIdx.x;
  if (i >= NGRAPH * D_OUT) return;
  int b = i / D_OUT, d = i % D_OUT;
  float c = cnt[b]; c = c > 1.f ? c : 1.f;
  emb[i] = pooled[i] / c + gbias[d];                     // bias folded into mean (constant)
}

// ---------- final classifier: [text | prem | hyp] @ Wc + bc ----------
__global__ void classifier_kernel(const float* __restrict__ text,
                                  const float* __restrict__ embp, const float* __restrict__ embh,
                                  const float* __restrict__ Wc, const float* __restrict__ bc,
                                  float* __restrict__ out) {
  int i = threadIdx.x;
  if (i >= NGRAPH * 3) return;
  int b = i / 3, c = i % 3;
  float acc = bc[c];
  const float* tf = text + b * D_IN;
  for (int k = 0; k < D_IN; ++k)  acc += tf[k] * Wc[k * 3 + c];
  const float* ep = embp + b * D_OUT;
  for (int k = 0; k < D_OUT; ++k) acc += ep[k] * Wc[(D_IN + k) * 3 + c];
  const float* eh = embh + b * D_OUT;
  for (int k = 0; k < D_OUT; ++k) acc += eh[k] * Wc[(D_IN + D_OUT + k) * 3 + c];
  out[i] = acc;
}

// ---------- one GATv2 side ----------
static void run_side(const float* x, const int* ei, const int* batch,
                     const bf16* Bl, const bf16* Br,
                     const float* blv, const float* brv, const float* att,
                     float* xl, float* xr, float* e_arr, unsigned* mkey,
                     float* denom, float* pooled, float* cnt,
                     const float* gbias, float* emb,
                     int n_nodes, int n_edges, hipStream_t stream) {
  int tot = n_edges + n_nodes;
  init_side_kernel<<<(n_nodes + 255) / 256, 256, 0, stream>>>(mkey, denom, pooled, cnt, n_nodes);
  gemm_xlxr_kernel<<<n_nodes / 16, 512, 0, stream>>>(x, Bl, Br, blv, brv, xl, xr);
  edge_logits_kernel<<<(tot + 7) / 8, 256, 0, stream>>>(ei, n_edges, n_nodes, xl, xr, att, e_arr, mkey);
  edge_exp_kernel<<<(tot + 255) / 256, 256, 0, stream>>>(ei, n_edges, n_nodes, mkey, e_arr, denom);
  int n4 = n_nodes * (D_OUT / 4);                        // zero h (reuses xr buffer)
  fill4_kernel<<<(n4 + 255) / 256, 256, 0, stream>>>((float4*)xr, n4);
  scatter_kernel<<<(tot + 7) / 8, 256, 0, stream>>>(ei, n_edges, n_nodes, e_arr, denom, xl, xr);
  pool_kernel<<<(n_nodes + 7) / 8, 256, 0, stream>>>(xr, batch, pooled, cnt, n_nodes);
  finalize_pool_kernel<<<(NGRAPH * D_OUT + 255) / 256, 256, 0, stream>>>(pooled, cnt, gbias, emb);
}

extern "C" void kernel_launch(void* const* d_in, const int* in_sizes, int n_in,
                              void* d_out, int out_size, void* d_ws, size_t ws_size,
                              hipStream_t stream) {
  const float* text   = (const float*)d_in[0];
  const float* px     = (const float*)d_in[1];
  const int*   pei    = (const int*)  d_in[2];
  const int*   pbatch = (const int*)  d_in[3];
  const float* hx     = (const float*)d_in[4];
  const int*   hei    = (const int*)  d_in[5];
  const int*   hbatch = (const int*)  d_in[6];
  const float* Wl     = (const float*)d_in[7];
  const float* blv    = (const float*)d_in[8];
  const float* Wr     = (const float*)d_in[9];
  const float* brv    = (const float*)d_in[10];
  const float* att    = (const float*)d_in[11];
  const float* gbias  = (const float*)d_in[12];
  const float* Wc     = (const float*)d_in[13];
  const float* bc     = (const float*)d_in[14];

  const int n_nodes = in_sizes[1] / D_IN;   // 65536
  const int n_edges = in_sizes[2] / 2;      // 262144

  char* ws = (char*)d_ws;
  bf16*     Bl     = (bf16*)    (ws + OFF_BL);
  bf16*     Br     = (bf16*)    (ws + OFF_BR);
  float*    e_arr  = (float*)   (ws + OFF_E);
  unsigned* mkey   = (unsigned*)(ws + OFF_MKEY);
  float*    denom  = (float*)   (ws + OFF_DENOM);
  float*    pooled = (float*)   (ws + OFF_POOLED);
  float*    cnt    = (float*)   (ws + OFF_CNT);
  float*    emb_p  = (float*)   (ws + OFF_EMB_P);
  float*    emb_h  = (float*)   (ws + OFF_EMB_H);
  float*    xl     = (float*)   (ws + OFF_XL);
  float*    xr     = (float*)   (ws + OFF_XR);

  pack_w_kernel<<<(KTILES * NTILES * 512 + 255) / 256, 256, 0, stream>>>(Wl, Wr, Bl, Br);

  run_side(px, pei, pbatch, Bl, Br, blv, brv, att,
           xl, xr, e_arr, mkey, denom, pooled, cnt, gbias, emb_p,
           n_nodes, n_edges, stream);
  run_side(hx, hei, hbatch, Bl, Br, blv, brv, att,
           xl, xr, e_arr, mkey, denom, pooled, cnt, gbias, emb_h,
           n_nodes, n_edges, stream);

  classifier_kernel<<<1, 64, 0, stream>>>(text, emb_p, emb_h, Wc, bc, (float*)d_out);
}
